// ModelComposition_54314156425443
// MI455X (gfx1250) — matrix operand order repaired and checked by reading, not compiled, not run on hardware
//
#include <hip/hip_runtime.h>
#include <hip/hip_bf16.h>
#include <math.h>

// ---------------------------------------------------------------------------
// Types for CDNA5 WMMA
// ---------------------------------------------------------------------------
typedef __attribute__((ext_vector_type(16))) _Float16 v16h;
typedef __attribute__((ext_vector_type(8)))  float    v8f;
typedef __attribute__((ext_vector_type(4)))  unsigned int v4u;

union Frag16 {
    v16h h;
    v4u  u[2];   // u[0] = halves 0..7, u[1] = halves 8..15
};

#define BATCH    16
#define NPTS     1024
#define MROWS    (BATCH * NPTS)   // 16384
#define CTXDIM   267
#define LATENT   256

// ---------------------------------------------------------------------------
// Pack fp32 weights -> f16, zero-padded to [Np x Kp] row-major
// ---------------------------------------------------------------------------
__global__ void pack_w_kernel(const float* __restrict__ W, _Float16* __restrict__ Wp,
                              int dout_, int din_, int Kp_, int total) {
    int idx = blockIdx.x * blockDim.x + threadIdx.x;
    if (idx >= total) return;
    int n = idx / Kp_;
    int k = idx - n * Kp_;
    float v = (n < dout_ && k < din_) ? W[n * din_ + k] : 0.0f;
    Wp[idx] = (_Float16)v;
}

// ---------------------------------------------------------------------------
// Pack current X state fp32 [16384 x 3] -> f16 [16384 x 32] (zero padded K)
// ---------------------------------------------------------------------------
__global__ void pack_x_kernel(const float* __restrict__ X, _Float16* __restrict__ Y) {
    int idx = blockIdx.x * blockDim.x + threadIdx.x;   // MROWS*32 threads
    int row = idx >> 5;
    int k   = idx & 31;
    float v = (k < 3) ? X[row * 3 + k] : 0.0f;
    Y[idx] = (_Float16)v;
}

// ---------------------------------------------------------------------------
// Per-step context gate/bias:  gate = sigmoid(ctx_full @ Wg^T + bg)
//                              cbias = ctx_full @ Wb^T
// ctx_full = [ctx(256) | beta,sin(beta),cos(beta) | label_emb(8)]
// ---------------------------------------------------------------------------
__global__ void ctx_gate_kernel(const float* __restrict__ ctx,
                                const int*   __restrict__ labels,
                                const float* __restrict__ lab_emb,
                                const float* __restrict__ Wg,
                                const float* __restrict__ Wb,
                                const float* __restrict__ bg,
                                float te0, float te1, float te2,
                                float* __restrict__ gate,
                                float* __restrict__ cbias,
                                int Np_, int dout_) {
    int idx = blockIdx.x * blockDim.x + threadIdx.x;
    if (idx >= BATCH * Np_) return;
    int b = idx / Np_;
    int n = idx - b * Np_;
    if (n >= dout_) { gate[idx] = 0.0f; cbias[idx] = 0.0f; return; }
    const float* c  = ctx + b * LATENT;
    const float* le = lab_emb + labels[b] * 8;
    const float* wg = Wg + n * CTXDIM;
    const float* wb = Wb + n * CTXDIM;
    float sg = bg[n];
    float sb = 0.0f;
    #pragma unroll 4
    for (int j = 0; j < LATENT; ++j) {
        sg += c[j] * wg[j];
        sb += c[j] * wb[j];
    }
    sg += te0 * wg[256] + te1 * wg[257] + te2 * wg[258];
    sb += te0 * wb[256] + te1 * wb[257] + te2 * wb[258];
    #pragma unroll
    for (int j = 0; j < 8; ++j) {
        sg += le[j] * wg[259 + j];
        sb += le[j] * wb[259 + j];
    }
    gate[idx]  = 1.0f / (1.0f + expf(-sg));
    cbias[idx] = sb;
}

// ---------------------------------------------------------------------------
// Fused context-gated GEMM layer, register-blocked 2(M) x NBLK(N) WMMA tiles:
//   Y[row,n] = gate[b,n] * (sum_k X[row,k]*W[n,k] + bias[n]) + cbias[b,n]
//   (+ReLU), f16 output for hidden layers, fp32 (first 3 cols) for the last.
//
// blockDim = 256 (8 waves). Each wave owns 2 consecutive 16-row M-tiles and
// the 16*NBLK-wide N-chunk selected by blockIdx.y. K swept by
// v_wmma_f32_16x16x32_f16 with f32 accumulation; A/B fragments are b128
// loads straight from row-major f16 buffers per ISA 7.12.2 layouts.
// grid = dim3(MROWS/256, NP/(16*NBLK)).
// ---------------------------------------------------------------------------
template <int KP, int NP, int NBLK, bool RELU, bool FINAL>
__global__ void gemm_ctx_kernel(const _Float16* __restrict__ Xin,
                                const _Float16* __restrict__ Wk,
                                const float* __restrict__ gate,
                                const float* __restrict__ cbias,
                                const float* __restrict__ bvec,
                                _Float16* __restrict__ Yf16,
                                float* __restrict__ Yf32) {
    constexpr int MBLK = 2;
    const int lane  = threadIdx.x & 31;
    const int wave  = threadIdx.x >> 5;
    const int mlo   = lane & 15;
    const int hi    = lane >> 4;                       // 0 or 1
    const int row0  = (blockIdx.x * 8 + wave) * (16 * MBLK);
    const int b_idx = row0 >> 10;                      // batch index (NPTS=1024)
    const int nc    = blockIdx.y * NBLK;               // first N-tile of this wave

    // A: lane -> M row, packed K (lanes 16..31 shifted by 8 in K)
    const _Float16* arow0 = Xin + (size_t)(row0 + mlo) * KP + hi * 8;
    const _Float16* arow1 = arow0 + (size_t)16 * KP;
    // B: lane -> N col, packed K (lanes 16..31 shifted by 8 in K)
    const _Float16* brow[NBLK];
    #pragma unroll
    for (int j = 0; j < NBLK; ++j)
        brow[j] = Wk + (size_t)((nc + j) * 16 + mlo) * KP + hi * 8;

    v8f acc[MBLK][NBLK] = {};

    for (int kt = 0; kt < KP; kt += 32) {
        Frag16 a[MBLK], b[NBLK];
        a[0].u[0] = *(const v4u*)(arow0 + kt);
        a[0].u[1] = *(const v4u*)(arow0 + kt + 16);
        a[1].u[0] = *(const v4u*)(arow1 + kt);
        a[1].u[1] = *(const v4u*)(arow1 + kt + 16);
        #pragma unroll
        for (int j = 0; j < NBLK; ++j) {
            b[j].u[0] = *(const v4u*)(brow[j] + kt);
            b[j].u[1] = *(const v4u*)(brow[j] + kt + 16);
        }
        #pragma unroll
        for (int i = 0; i < MBLK; ++i)
            #pragma unroll
            for (int j = 0; j < NBLK; ++j)
                acc[i][j] = __builtin_amdgcn_wmma_f32_16x16x32_f16(
                    /*neg_a=*/false, a[i].h, /*neg_b=*/false, b[j].h,
                    /*c_mod=*/(short)0, acc[i][j],
                    /*reuse_a=*/false, /*reuse_b=*/false);
    }

    // Epilogue: each lane owns one output column per N-tile, 8 rows per M-tile.
    #pragma unroll
    for (int j = 0; j < NBLK; ++j) {
        const int ncol = (nc + j) * 16 + mlo;
        const float g  = gate [b_idx * NP + ncol];
        const float cb = cbias[b_idx * NP + ncol];
        const float bs = FINAL ? ((ncol < 3) ? bvec[ncol] : 0.0f) : bvec[ncol];
        #pragma unroll
        for (int i = 0; i < MBLK; ++i) {
            #pragma unroll
            for (int r = 0; r < 8; ++r) {
                const int row = row0 + i * 16 + r + hi * 8;  // D layout: VGPR r -> M
                float y = g * (acc[i][j][r] + bs) + cb;
                if (RELU) y = fmaxf(y, 0.0f);
                if (FINAL) {
                    if (ncol < 3) Yf32[(size_t)row * 3 + ncol] = y;
                } else {
                    Yf16[(size_t)row * NP + ncol] = (_Float16)y;
                }
            }
        }
    }
}

// ---------------------------------------------------------------------------
// Reverse-diffusion state update: X = (X - c0*eps)*c1 + sigma*z
// ---------------------------------------------------------------------------
__global__ void x_update_kernel(const float* __restrict__ Xc,
                                const float* __restrict__ eps,
                                const float* __restrict__ z,
                                float* __restrict__ Xo,
                                float c0, float c1, float sigma, int n) {
    int i = blockIdx.x * blockDim.x + threadIdx.x;
    if (i < n) Xo[i] = (Xc[i] - c0 * eps[i]) * c1 + sigma * z[i];
}

// ---------------------------------------------------------------------------
// Host orchestration
// ---------------------------------------------------------------------------
extern "C" void kernel_launch(void* const* d_in, const int* in_sizes, int n_in,
                              void* d_out, int out_size, void* d_ws, size_t ws_size,
                              hipStream_t stream) {
    (void)in_sizes; (void)n_in; (void)out_size; (void)ws_size;

    // ---- input pointers (setup_inputs dict order, params flattened in-order)
    const int*   labels  = (const int*)  d_in[0];
    const float* ctx     = (const float*)d_in[1];
    const float* X_init  = (const float*)d_in[2];
    const float* noise   = (const float*)d_in[3];
    const float* Wf[6], *bf[6], *Wbf[6], *Wgf[6], *bgf[6];
    for (int l = 0; l < 6; ++l) {
        Wf[l]  = (const float*)d_in[4 + 5 * l + 0];
        bf[l]  = (const float*)d_in[4 + 5 * l + 1];
        Wbf[l] = (const float*)d_in[4 + 5 * l + 2];
        Wgf[l] = (const float*)d_in[4 + 5 * l + 3];
        bgf[l] = (const float*)d_in[4 + 5 * l + 4];
    }
    const float* lab_emb = (const float*)d_in[34];

    static const int din_[6]  = {3, 128, 256, 512, 256, 128};
    static const int dout_[6] = {128, 256, 512, 256, 128, 3};
    static const int Kp_[6]   = {32, 128, 256, 512, 256, 128};
    static const int Np_[6]   = {128, 256, 512, 256, 128, 16};

    // ---- workspace carve (256B aligned)
    char* p = (char*)d_ws;
    auto alloc = [&](size_t bytes) -> char* {
        char* r = p;
        p += (bytes + 255) & ~(size_t)255;
        return r;
    };
    _Float16* wpk[6];
    float *gate[6], *cbias[6];
    for (int l = 0; l < 6; ++l) wpk[l]   = (_Float16*)alloc((size_t)Np_[l] * Kp_[l] * 2);
    for (int l = 0; l < 6; ++l) gate[l]  = (float*)alloc((size_t)BATCH * Np_[l] * 4);
    for (int l = 0; l < 6; ++l) cbias[l] = (float*)alloc((size_t)BATCH * Np_[l] * 4);
    _Float16* XA  = (_Float16*)alloc((size_t)MROWS * 256 * 2);  // widths 32/256
    _Float16* XB  = (_Float16*)alloc((size_t)MROWS * 512 * 2);  // widths 128/512
    float*    eps = (float*)alloc((size_t)MROWS * 3 * 4);
    float*    Xst = (float*)alloc((size_t)MROWS * 3 * 4);

    // ---- one-time weight packing
    for (int l = 0; l < 6; ++l) {
        int total = Np_[l] * Kp_[l];
        pack_w_kernel<<<(total + 255) / 256, 256, 0, stream>>>(
            Wf[l], wpk[l], dout_[l], din_[l], Kp_[l], total);
    }

    // ---- host-side schedule constants (pure CPU math, capture-safe)
    const int T = 20;
    double betas[20], alphas[20], aprod[20], aprod_prev[20];
    for (int t = 0; t < T; ++t) {
        betas[t]  = 1e-4 + (0.05 - 1e-4) * (double)t / (double)(T - 1);
        alphas[t] = 1.0 - betas[t];
    }
    double cp = 1.0;
    for (int t = 0; t < T; ++t) { aprod_prev[t] = cp; cp *= alphas[t]; aprod[t] = cp; }

    const int nelem   = MROWS * 3;
    const int mblocks = MROWS / 256;   // 64 blocks of 8 waves x 2 M-tiles

    for (int s = 0; s < T; ++s) {
        const int t = T - 1 - s;                       // reversed time order
        const float beta  = (float)betas[t];
        const float c0    = (float)(betas[t] / sqrt(1.0 - aprod[t]));
        const float c1    = (float)(1.0 / sqrt(alphas[t]));
        const float sigma = (float)sqrt(betas[t] * (1.0 - aprod_prev[t]) / (1.0 - aprod[t]));
        const float te0 = beta, te1 = sinf(beta), te2 = cosf(beta);
        const float* z    = noise + (size_t)s * nelem;
        const float* Xcur = (s == 0) ? X_init : Xst;
        float*       Xout = (s == T - 1) ? (float*)d_out : Xst;

        // pack current state into f16 layer-0 input (K padded 3 -> 32)
        pack_x_kernel<<<(MROWS * 32) / 256, 256, 0, stream>>>(Xcur, XA);

        // per-layer gate/bias from context (tiny)
        for (int l = 0; l < 6; ++l) {
            int total = BATCH * Np_[l];
            ctx_gate_kernel<<<(total + 255) / 256, 256, 0, stream>>>(
                ctx, labels, lab_emb, Wgf[l], Wbf[l], bgf[l],
                te0, te1, te2, gate[l], cbias[l], Np_[l], dout_[l]);
        }

        // 6 fused WMMA GEMM layers, ping-ponging XA/XB
        // grid = (M blocks, N chunks of 16*NBLK columns)
        gemm_ctx_kernel<32, 128, 4, true, false>
            <<<dim3(mblocks, 128 / 64), 256, 0, stream>>>(
                XA, wpk[0], gate[0], cbias[0], bf[0], XB, nullptr);
        gemm_ctx_kernel<128, 256, 4, true, false>
            <<<dim3(mblocks, 256 / 64), 256, 0, stream>>>(
                XB, wpk[1], gate[1], cbias[1], bf[1], XA, nullptr);
        gemm_ctx_kernel<256, 512, 4, true, false>
            <<<dim3(mblocks, 512 / 64), 256, 0, stream>>>(
                XA, wpk[2], gate[2], cbias[2], bf[2], XB, nullptr);
        gemm_ctx_kernel<512, 256, 4, true, false>
            <<<dim3(mblocks, 256 / 64), 256, 0, stream>>>(
                XB, wpk[3], gate[3], cbias[3], bf[3], XA, nullptr);
        gemm_ctx_kernel<256, 128, 4, true, false>
            <<<dim3(mblocks, 128 / 64), 256, 0, stream>>>(
                XA, wpk[4], gate[4], cbias[4], bf[4], XB, nullptr);
        gemm_ctx_kernel<128, 16, 1, false, true>
            <<<dim3(mblocks, 1), 256, 0, stream>>>(
                XB, wpk[5], gate[5], cbias[5], bf[5], nullptr, eps);

        // reverse-diffusion update
        x_update_kernel<<<(nelem + 255) / 256, 256, 0, stream>>>(
            Xcur, eps, z, Xout, c0, c1, sigma, nelem);
    }
}
